// ARANSMTSllm_73409581023282
// MI455X (gfx1250) — matrix operand order repaired and checked by reading, not compiled, not run on hardware
//
#include <hip/hip_runtime.h>
#include <hip/hip_bf16.h>
#include <stdint.h>

typedef __attribute__((ext_vector_type(16))) _Float16 v16h;
typedef __attribute__((ext_vector_type(8)))  _Float16 v8h;
typedef __attribute__((ext_vector_type(8)))  float    v8f;

#define B_BATCH 32
#define LHIST   512
#define LKB     608
#define NKB     50000
#define NTILES  (NKB / 16)                 // 3125 column tiles
#define WAVES_PER_WG 8
#define WG_THREADS   (WAVES_PER_WG * 32)   // wave32 => 8 waves
#define NWG ((NTILES + WAVES_PER_WG - 1) / WAVES_PER_WG)
#define PLANE (B_BATCH * LHIST)            // 16384 f16 per plane

static __device__ __forceinline__ v8f wmma_f16(v16h a, v16h b, v8f c) {
  // D = A(16x32 f16) * B(32x16 f16) + C(16x16 f32)
  return __builtin_amdgcn_wmma_f32_16x16x32_f16(false, a, false, b, (short)0, c,
                                                false, false);
}

static __device__ __forceinline__ unsigned sortable(float f) {
  unsigned u = __float_as_uint(f);
  return (u & 0x80000000u) ? ~u : (u | 0x80000000u);
}

__global__ void init_ws_kernel(unsigned long long* ws, int stride) {
  ws[(size_t)threadIdx.x * stride] = ~0ULL;
}

__global__ __launch_bounds__(WG_THREADS)
void argmin_kernel(const float* __restrict__ x, const float* __restrict__ kb,
                   unsigned long long* __restrict__ ws, int slot_stride) {
  // hi plane + lo plane of x as f16, xor-swizzled rows: 64 KB LDS
  __shared__ __attribute__((aligned(16))) _Float16 sh[2 * PLANE];

  const int tid = threadIdx.x;
  // Stage x (32 x 512 f32) -> f16 hi/lo split in LDS
  for (int i = tid; i < PLANE; i += WG_THREADS) {
    int m = i >> 9, k = i & (LHIST - 1);
    float v = x[i];
    _Float16 h = (_Float16)v;
    _Float16 l = (_Float16)(v - (float)h);
    int pos = (m << 9) + (k ^ ((m & 15) << 3));  // swizzle bits 3..6 of k
    sh[pos]         = h;
    sh[PLANE + pos] = l;
  }
  __syncthreads();

  const int wave = tid >> 5;
  const int lane = tid & 31;
  const int task = blockIdx.x * WAVES_PER_WG + wave;   // one N-tile per wave
  if (task >= NTILES) return;

  const int col  = lane & 15;            // B-matrix column (N within tile)
  const int half = lane >> 4;            // K-half selector
  const int row  = task * 16 + col;      // global KB row handled by this lane
  const float* __restrict__ kbrow = kb + (size_t)row * LKB + 16 * half;

  const int swz    = col << 3;
  const int baseA0 = col << 9;           // A rows 0..15  (batches 0..15)
  const int baseA1 = (col + 16) << 9;    // A rows 16..31 (batches 16..31)

  v8f acc0 = {0.f, 0.f, 0.f, 0.f, 0.f, 0.f, 0.f, 0.f};
  v8f acc1 = acc0;
  float sumsq = 0.f;

#pragma unroll 2
  for (int s = 0; s < 16; ++s) {          // K = 512 in steps of 32
    // ---- B tile: 16 contiguous f32 per lane (one 128B line per row pair)
    const float4* p = (const float4*)(kbrow + 32 * s);
    float4 f0 = p[0], f1 = p[1], f2 = p[2], f3 = p[3];
    float fv[16];
    fv[0]=f0.x; fv[1]=f0.y; fv[2]=f0.z;  fv[3]=f0.w;
    fv[4]=f1.x; fv[5]=f1.y; fv[6]=f1.z;  fv[7]=f1.w;
    fv[8]=f2.x; fv[9]=f2.y; fv[10]=f2.z; fv[11]=f2.w;
    fv[12]=f3.x; fv[13]=f3.y; fv[14]=f3.z; fv[15]=f3.w;

    v16h bh, bl;
#pragma unroll
    for (int e = 0; e < 16; ++e) {
      float v = fv[e];
      sumsq = __builtin_fmaf(v, v, sumsq);   // exact f32 row norms
      _Float16 h = (_Float16)v;
      bh[e] = h;
      bl[e] = (_Float16)(v - (float)h);      // f32 = hi + lo split
    }

    // ---- A fragments from LDS (contiguous k = 16*half + e, same order as B)
    const int k0 = 32 * s + 16 * half;
    const int i1 = k0 ^ swz;
    const int i2 = (k0 + 8) ^ swz;
    v8h h0a = *(const v8h*)(sh + baseA0 + i1);
    v8h h0b = *(const v8h*)(sh + baseA0 + i2);
    v8h l0a = *(const v8h*)(sh + PLANE + baseA0 + i1);
    v8h l0b = *(const v8h*)(sh + PLANE + baseA0 + i2);
    v8h h1a = *(const v8h*)(sh + baseA1 + i1);
    v8h h1b = *(const v8h*)(sh + baseA1 + i2);
    v8h l1a = *(const v8h*)(sh + PLANE + baseA1 + i1);
    v8h l1b = *(const v8h*)(sh + PLANE + baseA1 + i2);
    v16h ah0 = __builtin_shufflevector(h0a, h0b, 0,1,2,3,4,5,6,7,8,9,10,11,12,13,14,15);
    v16h al0 = __builtin_shufflevector(l0a, l0b, 0,1,2,3,4,5,6,7,8,9,10,11,12,13,14,15);
    v16h ah1 = __builtin_shufflevector(h1a, h1b, 0,1,2,3,4,5,6,7,8,9,10,11,12,13,14,15);
    v16h al1 = __builtin_shufflevector(l1a, l1b, 0,1,2,3,4,5,6,7,8,9,10,11,12,13,14,15);

    // cross ~= xh*bh + xh*bl + xl*bh  (xl*bl ~ 2^-22, dropped)
    acc0 = wmma_f16(ah0, bh, acc0);
    acc1 = wmma_f16(ah1, bh, acc1);
    acc0 = wmma_f16(ah0, bl, acc0);
    acc1 = wmma_f16(ah1, bl, acc1);
    acc0 = wmma_f16(al0, bh, acc0);
    acc1 = wmma_f16(al1, bh, acc1);
  }

  // kb_sq for this lane's column: combine the two K-halves
  float ss = sumsq + __shfl_xor(sumsq, 16, 32);
  const unsigned ncol = (unsigned)row;    // global n for this lane's column

#pragma unroll
  for (int r = 0; r < 8; ++r) {
    // C/D layout: lanes 0-15 -> M=r, lanes 16-31 -> M=r+8, N = lane&15
    float sc0 = __builtin_fmaf(-2.f, acc0[r], ss);  // kb_sq - 2*cross
    float sc1 = __builtin_fmaf(-2.f, acc1[r], ss);
    unsigned long long k0 = ((unsigned long long)sortable(sc0) << 32) | ncol;
    unsigned long long k1 = ((unsigned long long)sortable(sc1) << 32) | ncol;
#pragma unroll
    for (int d = 1; d < 16; d <<= 1) {    // min over 16 columns (per half)
      unsigned long long o0 = __shfl_xor(k0, d, 32);
      unsigned long long o1 = __shfl_xor(k1, d, 32);
      k0 = o0 < k0 ? o0 : k0;
      k1 = o1 < k1 ? o1 : k1;
    }
    if (col == 0) {
      int m = r + 8 * half;
      atomicMin(ws + (size_t)m * slot_stride, k0);
      atomicMin(ws + (size_t)(m + 16) * slot_stride, k1);
    }
  }
}

__global__ void gather_kernel(const float* __restrict__ kb,
                              const unsigned long long* __restrict__ ws,
                              float* __restrict__ out, int slot_stride) {
  const int b = blockIdx.x;
  const unsigned n = (unsigned)(ws[(size_t)b * slot_stride] & 0xFFFFFFFFULL);
  const float* src = kb + (size_t)n * LKB;
  for (int j = threadIdx.x; j < LKB; j += blockDim.x)
    out[b * LKB + j] = src[j];
}

extern "C" void kernel_launch(void* const* d_in, const int* in_sizes, int n_in,
                              void* d_out, int out_size, void* d_ws, size_t ws_size,
                              hipStream_t stream) {
  (void)in_sizes; (void)n_in; (void)out_size;
  const float* x  = (const float*)d_in[0];   // (32, 512, 1) f32
  const float* kb = (const float*)d_in[1];   // (50000, 608, 1) f32
  float* out = (float*)d_out;                // (32, 608, 1) f32
  unsigned long long* ws = (unsigned long long*)d_ws;

  // Pad argmin slots to 256B to spread atomics across L2 channels.
  const int slot_stride = (ws_size >= (size_t)B_BATCH * 32 * 8) ? 32 : 1;

  init_ws_kernel<<<1, B_BATCH, 0, stream>>>(ws, slot_stride);
  argmin_kernel<<<NWG, WG_THREADS, 0, stream>>>(x, kb, ws, slot_stride);
  gather_kernel<<<B_BATCH, 256, 0, stream>>>(kb, ws, out, slot_stride);
}
